// SelfAttention_Dynamicv2_32916629357180
// MI455X (gfx1250) — matrix operand order repaired and checked by reading, not compiled, not run on hardware
//
#include <hip/hip_runtime.h>
#include <hip/hip_bf16.h>

#define B_  8
#define T_  16
#define N_  256
#define C_  1024
#define H_  16
#define K_  8
#define BT_ 128
#define HD_ 64

typedef __attribute__((ext_vector_type(16))) _Float16 v16h;
typedef __attribute__((ext_vector_type(8)))  _Float16 v8h;
typedef __attribute__((ext_vector_type(2)))  _Float16 h2;
typedef __attribute__((ext_vector_type(8)))  float    v8f;

// ---------------- CDNA5 async global->LDS copy (ASYNCcnt tracked) ----------
// LDS[lds_off .. +15] = MEM[g .. +15]   (per active lane)
__device__ inline void async_copy_b128(const void* g, unsigned lds_off) {
  asm volatile("global_load_async_to_lds_b128 %0, %1, off"
               :: "v"(lds_off), "v"(g) : "memory");
}
__device__ inline void async_wait0() {
  asm volatile("s_wait_asynccnt 0x0" ::: "memory");
}
__device__ inline unsigned lds_off_of(const void* p) {
  return (unsigned)(uintptr_t)p;      // generic LDS addr[31:0] == LDS offset
}

// ---------------- WMMA fragment loaders (ISA 7.12.2 layouts) ----------------
// A-matrix 16x32 f16: lane m=L&15, g=L>>4 ; VGPR j (elements 2j,2j+1):
//   j<4: K = 2j + 8g ; j>=4: K = 2j + 8 + 8g
__device__ inline v16h frag_a(const _Float16* base, int ld) {
  int L = threadIdx.x & 31;
  const _Float16* row = base + (L & 15) * ld;
  int g8 = (L >> 4) * 8;
  v16h f;
#pragma unroll
  for (int j = 0; j < 8; ++j) {
    int kb = ((j & 4) ? (2 * j + 8) : (2 * j)) + g8;
    h2 p = *(const h2*)(row + kb);
    f[2 * j]     = p[0];
    f[2 * j + 1] = p[1];
  }
  return f;
}

// B-matrix 32x16 f16 (stored transposed in LDS as [n][k]):
// lane n=L&15, g=L>>4 ; element e holds K = 8g + e (e<8), 16 + 8g + (e-8) (e>=8)
__device__ inline v16h frag_b(const _Float16* base, int ld) {
  int L = threadIdx.x & 31;
  const _Float16* row = base + (L & 15) * ld + (L >> 4) * 8;
  v16h f;
#pragma unroll
  for (int j = 0; j < 8; ++j) {
    h2 p = *(const h2*)(row + ((j & 4) ? (2 * j + 8) : (2 * j)));
    f[2 * j]     = p[0];
    f[2 * j + 1] = p[1];
  }
  return f;
}

__device__ inline v8f wmma_f16(v16h a, v16h b, v8f c) {
  return __builtin_amdgcn_wmma_f32_16x16x32_f16(false, a, false, b, (short)0, c,
                                                false, false);
}

// ---------------- Stage 0: expert sums over axis 1 ----------------
__global__ void expert_sum_kernel(const float* __restrict__ qe,
                                  const float* __restrict__ ke,
                                  float* __restrict__ qsum,
                                  float* __restrict__ ksum) {
  int gid = blockIdx.x * 256 + threadIdx.x;   // 0 .. 2*K*C-1
  int which = gid >= K_ * C_;
  int kc = which ? gid - K_ * C_ : gid;
  int k = kc / C_, c = kc % C_;
  const float* src = (which ? ke : qe) + (size_t)k * C_ * C_ + c;
  float s = 0.f;
  for (int i = 0; i < C_; ++i) s += src[(size_t)i * C_];
  (which ? ksum : qsum)[kc] = s;
}

// ---------------- Stage 0b: Wc = wv @ v_w ; bc = wv@v_b + bv ----------------
__global__ void combine_w_kernel(const float* __restrict__ in_w,
                                 const float* __restrict__ v_w,
                                 float* __restrict__ Wc) {
  int gid = blockIdx.x * 256 + threadIdx.x;   // j*C + c
  int j = gid >> 10, c = gid & (C_ - 1);
  const float* wv = in_w + (size_t)2 * C_ * C_ + (size_t)j * C_;
  float s = 0.f;
  for (int i = 0; i < C_; ++i) s += wv[i] * v_w[(size_t)i * C_ + c];
  Wc[gid] = s;
}

__global__ void combine_b_kernel(const float* __restrict__ in_w,
                                 const float* __restrict__ v_b,
                                 const float* __restrict__ in_b,
                                 float* __restrict__ bc) {
  int j = blockIdx.x * 256 + threadIdx.x;
  const float* wv = in_w + (size_t)2 * C_ * C_ + (size_t)j * C_;
  float s = in_b[2 * C_ + j];
  for (int i = 0; i < C_; ++i) s += wv[i] * v_b[i];
  bc[j] = s;
}

// ---------------- Stage 1: dyn softmax + q/k scales ----------------
__global__ __launch_bounds__(256) void dyn_scale_kernel(
    const float* __restrict__ y, const float* __restrict__ dyn_w,
    const float* __restrict__ dyn_b, const float* __restrict__ qsum,
    const float* __restrict__ ksum, float* __restrict__ qscale,
    float* __restrict__ kscale) {
  int bn = blockIdx.x;           // b*N + n
  int t = threadIdx.x;
  __shared__ float sP[K_][256];
  __shared__ float sDyn[K_];
  float acc[K_];
#pragma unroll
  for (int k = 0; k < K_; ++k) acc[k] = 0.f;
  const float* yrow = y + (size_t)bn * C_;
  for (int c = t; c < C_; c += 256) {
    float yv = yrow[c];
#pragma unroll
    for (int k = 0; k < K_; ++k) acc[k] += yv * dyn_w[k * C_ + c];
  }
#pragma unroll
  for (int k = 0; k < K_; ++k) sP[k][t] = acc[k];
  __syncthreads();
  for (int off = 128; off > 0; off >>= 1) {
    if (t < off) {
#pragma unroll
      for (int k = 0; k < K_; ++k) sP[k][t] += sP[k][t + off];
    }
    __syncthreads();
  }
  if (t == 0) {
    float lg[K_], mx = -3.4e38f;
#pragma unroll
    for (int k = 0; k < K_; ++k) { lg[k] = sP[k][0] + dyn_b[k]; mx = fmaxf(mx, lg[k]); }
    float s = 0.f;
#pragma unroll
    for (int k = 0; k < K_; ++k) { lg[k] = __expf(lg[k] - mx); s += lg[k]; }
    float inv = 1.f / s;
#pragma unroll
    for (int k = 0; k < K_; ++k) sDyn[k] = lg[k] * inv;
  }
  __syncthreads();
  float d[K_];
#pragma unroll
  for (int k = 0; k < K_; ++k) d[k] = sDyn[k];
  for (int c = t; c < C_; c += 256) {
    float qs = 0.f, ks = 0.f;
#pragma unroll
    for (int k = 0; k < K_; ++k) {
      qs += d[k] * qsum[k * C_ + c];
      ks += d[k] * ksum[k * C_ + c];
    }
    size_t o = (size_t)bn * C_ + c;
    qscale[o] = qs;
    kscale[o] = ks;
  }
}

// ---------------- Stage 2: WMMA projection GEMM ----------------
// out[m, j] = A[m, :] . W[j, :] + bias[j]
// mode 0/1: A = x * scale(b,n,:)   -> f16 out
// mode 2  : A = x                  -> f16 out
// mode 3  : A = ctx(f16), out f32 = gemm + bias + resid
#define BM 64
#define BN 128
#define BK 32
#define LDA 40

// grid: (C/BN, M/BM)  -- n varies fastest so the x-strip stays L2-resident
__global__ __launch_bounds__(256) void proj_gemm_kernel(
    const float* __restrict__ Af, const _Float16* __restrict__ Ah,
    const float* __restrict__ scale, const float* __restrict__ W,
    const float* __restrict__ bias, _Float16* __restrict__ outH,
    float* __restrict__ outF, const float* __restrict__ resid, int mode) {
  __shared__ _Float16 sA[BM * LDA];
  __shared__ _Float16 sB[BN * LDA];
  int n0 = blockIdx.x * BN;
  int m0 = blockIdx.y * BM;
  int t = threadIdx.x;
  int w = t >> 5;
  int rt = w & 3;        // 16-row tile within BM
  int cg = w >> 2;       // 64-col group within BN
  v8f zero = {};
  v8f acc[4] = {zero, zero, zero, zero};
  int arow = t >> 2;            // 0..63
  int acol = (t & 3) * 8;       // 0,8,16,24
  int brow = t >> 1;            // 0..127
  int bcol = (t & 1) * 16;      // 0 or 16

  for (int k0 = 0; k0 < C_; k0 += BK) {
    __syncthreads();
    {  // A tile
      int m = m0 + arow;
      if (mode == 3) {
        const _Float16* src = Ah + (size_t)m * C_ + k0 + acol;
        async_copy_b128(src, lds_off_of(&sA[arow * LDA + acol]));
      } else if (mode <= 1) {
        const float* src = Af + (size_t)m * C_ + k0 + acol;
        int b = m >> 12;              // / (T*N)
        int n = m & (N_ - 1);
        const float* sc = scale + ((size_t)b * N_ + n) * C_ + k0 + acol;
#pragma unroll
        for (int u = 0; u < 8; ++u)
          sA[arow * LDA + acol + u] = (_Float16)(src[u] * sc[u]);
      } else {
        const float* src = Af + (size_t)m * C_ + k0 + acol;
#pragma unroll
        for (int u = 0; u < 8; ++u) sA[arow * LDA + acol + u] = (_Float16)src[u];
      }
    }
    {  // B tile: sB[n][k] = W[n0+n][k0+k]
      const float* srcW = W + (size_t)(n0 + brow) * C_ + k0 + bcol;
#pragma unroll
      for (int u = 0; u < 16; ++u) sB[brow * LDA + bcol + u] = (_Float16)srcW[u];
    }
    if (mode == 3) async_wait0();
    __syncthreads();
    v16h a = frag_a(&sA[(rt * 16) * LDA], LDA);
    v16h bf[4];
#pragma unroll
    for (int j = 0; j < 4; ++j)
      bf[j] = frag_b(&sB[(cg * 64 + j * 16) * LDA], LDA);
#pragma unroll
    for (int j = 0; j < 4; ++j) acc[j] = wmma_f16(a, bf[j], acc[j]);
  }
  int lane = t & 31, idx = lane & 15, g = lane >> 4;
#pragma unroll
  for (int j = 0; j < 4; ++j) {
    int col = n0 + cg * 64 + j * 16 + idx;
    float bval = bias[col];
#pragma unroll
    for (int r = 0; r < 8; ++r) {
      int row = m0 + rt * 16 + r + 8 * g;
      float v = acc[j][r] + bval;
      size_t o = (size_t)row * C_ + col;
      if (mode == 3) outF[o] = v + resid[o];
      else           outH[o] = (_Float16)v;
    }
  }
}

// ---------------- Stage 3: fused attention per (bt, 32-row q tile) ----------
__global__ __launch_bounds__(256) void attention_kernel(
    const _Float16* __restrict__ qp, const _Float16* __restrict__ kp,
    const _Float16* __restrict__ vp, const unsigned char* __restrict__ mask,
    _Float16* __restrict__ ctx, float* __restrict__ avg) {
  __shared__ _Float16 sQ[32 * 72];
  __shared__ _Float16 sKV[256 * 72];     // aliased as vp^T [64 x 264]
  __shared__ _Float16 sATT[32 * 264];
  __shared__ float sMax[32][4];
  __shared__ float sSum[32][4];

  int bt = blockIdx.x >> 3;
  int qt = blockIdx.x & 7;
  int q0 = qt * 32;
  int t = threadIdx.x;
  int w = t >> 5;
  int lane = t & 31, idx = lane & 15, g = lane >> 4;
  int rt = w & 1;        // 16-row tile of the 32 q rows
  int cb = w >> 1;       // 64-col block of the 256 k cols
  const size_t baseQ = (size_t)bt * N_ * C_;

  float mpen[4];
#pragma unroll
  for (int j = 0; j < 4; ++j) {
    int col = cb * 64 + j * 16 + idx;
    mpen[j] = mask[(size_t)bt * N_ + col] ? 0.f : 1.f;   // 1 => masked out
  }
  float avgacc[4][8];
#pragma unroll
  for (int j = 0; j < 4; ++j)
#pragma unroll
    for (int r = 0; r < 8; ++r) avgacc[j][r] = 0.f;

  for (int h = 0; h < H_; ++h) {
    __syncthreads();
    {  // q tile 32x64 : async global -> LDS
      int r = t >> 3, d0 = (t & 7) * 8;
      async_copy_b128(qp + baseQ + (size_t)(q0 + r) * C_ + h * HD_ + d0,
                      lds_off_of(&sQ[r * 72 + d0]));
    }
    {  // k rows 256x64 : async global -> LDS
      int c_ = t & 7, r0 = t >> 3;
#pragma unroll
      for (int i = 0; i < 8; ++i) {
        int kk = r0 + 32 * i;
        async_copy_b128(kp + baseQ + (size_t)kk * C_ + h * HD_ + c_ * 8,
                        lds_off_of(&sKV[kk * 72 + c_ * 8]));
      }
    }
    async_wait0();
    __syncthreads();

    // scores = q . k^T / 8
    v8f zero = {};
    v8f sacc[4] = {zero, zero, zero, zero};
#pragma unroll
    for (int ks = 0; ks < 2; ++ks) {
      v16h a = frag_a(&sQ[(rt * 16) * 72 + ks * 32], 72);
      v16h bf[4];
#pragma unroll
      for (int j = 0; j < 4; ++j)
        bf[j] = frag_b(&sKV[(cb * 64 + j * 16) * 72 + ks * 32], 72);
#pragma unroll
      for (int j = 0; j < 4; ++j) sacc[j] = wmma_f16(a, bf[j], sacc[j]);
    }
    float val[4][8];
#pragma unroll
    for (int j = 0; j < 4; ++j)
#pragma unroll
      for (int r = 0; r < 8; ++r)
        val[j][r] = (mpen[j] > 0.f) ? -3.4028235e38f : sacc[j][r] * 0.125f;

    // row max across this wave's 64 cols, then across waves
    float rm[8];
#pragma unroll
    for (int r = 0; r < 8; ++r) {
      float m = val[0][r];
#pragma unroll
      for (int j = 1; j < 4; ++j) m = fmaxf(m, val[j][r]);
#pragma unroll
      for (int mk = 8; mk >= 1; mk >>= 1) m = fmaxf(m, __shfl_xor(m, mk, 32));
      rm[r] = m;
    }
    if (idx == 0) {
#pragma unroll
      for (int r = 0; r < 8; ++r) sMax[rt * 16 + r + 8 * g][cb] = rm[r];
    }
    __syncthreads();
    float rs[8];
#pragma unroll
    for (int r = 0; r < 8; ++r) {
      int row = rt * 16 + r + 8 * g;
      float m = fmaxf(fmaxf(sMax[row][0], sMax[row][1]),
                      fmaxf(sMax[row][2], sMax[row][3]));
      float s = 0.f;
#pragma unroll
      for (int j = 0; j < 4; ++j) { val[j][r] = __expf(val[j][r] - m); s += val[j][r]; }
#pragma unroll
      for (int mk = 8; mk >= 1; mk >>= 1) s += __shfl_xor(s, mk, 32);
      rs[r] = s;
    }
    if (idx == 0) {
#pragma unroll
      for (int r = 0; r < 8; ++r) sSum[rt * 16 + r + 8 * g][cb] = rs[r];
    }
    __syncthreads();
#pragma unroll
    for (int r = 0; r < 8; ++r) {
      int row = rt * 16 + r + 8 * g;
      float inv = 1.f / (sSum[row][0] + sSum[row][1] + sSum[row][2] + sSum[row][3]);
#pragma unroll
      for (int j = 0; j < 4; ++j) {
        float a_ = val[j][r] * inv;
        avgacc[j][r] += a_ * (1.f / H_);
        sATT[row * 264 + cb * 64 + j * 16 + idx] = (_Float16)a_;
      }
    }
    __syncthreads();      // sATT complete, kp reads done

    {  // vp transposed into aliased buffer: sVT[d][kk] (transpose => no async)
      _Float16* sVT = sKV;
      int c_ = t & 7, r0 = t >> 3;
#pragma unroll
      for (int i = 0; i < 8; ++i) {
        int kk = r0 + 32 * i;
        v8h v = *(const v8h*)(vp + baseQ + (size_t)kk * C_ + h * HD_ + c_ * 8);
#pragma unroll
        for (int u = 0; u < 8; ++u) sVT[(c_ * 8 + u) * 264 + kk] = v[u];
      }
    }
    __syncthreads();

    // ctx = attn @ vp : one 16x16 tile per wave, K = 256
    int rt2 = w & 1, ct = w >> 1;
    v8f cacc = {};
#pragma unroll
    for (int ks = 0; ks < 8; ++ks) {
      v16h a  = frag_a(&sATT[(rt2 * 16) * 264 + ks * 32], 264);
      v16h bf = frag_b(&sKV[(ct * 16) * 264 + ks * 32], 264);  // sVT alias
      cacc = wmma_f16(a, bf, cacc);
    }
#pragma unroll
    for (int r = 0; r < 8; ++r) {
      int row = q0 + rt2 * 16 + r + 8 * g;
      int col = h * HD_ + ct * 16 + idx;
      ctx[baseQ + (size_t)row * C_ + col] = (_Float16)cacc[r];
    }
  }

  // attn_avg (mean over heads), single-owner writes
#pragma unroll
  for (int j = 0; j < 4; ++j)
#pragma unroll
    for (int r = 0; r < 8; ++r) {
      int row = q0 + rt * 16 + r + 8 * g;
      int col = cb * 64 + j * 16 + idx;
      avg[(size_t)bt * N_ * N_ + (size_t)row * N_ + col] = avgacc[j][r];
    }
}

// ---------------- Stage 4: in-place LayerNorm ----------------
__global__ __launch_bounds__(256) void layernorm_kernel(
    float* __restrict__ hbuf, const float* __restrict__ gamma,
    const float* __restrict__ beta) {
  int m = blockIdx.x;
  int t = threadIdx.x;
  __shared__ float s1[256], s2[256];
  float* row = hbuf + (size_t)m * C_;
  float v[4], a = 0.f, b = 0.f;
#pragma unroll
  for (int i = 0; i < 4; ++i) {
    v[i] = row[t + 256 * i];
    a += v[i];
    b += v[i] * v[i];
  }
  s1[t] = a; s2[t] = b;
  __syncthreads();
  for (int off = 128; off > 0; off >>= 1) {
    if (t < off) { s1[t] += s1[t + off]; s2[t] += s2[t + off]; }
    __syncthreads();
  }
  float mu  = s1[0] * (1.f / C_);
  float var = s2[0] * (1.f / C_) - mu * mu;
  float inv = rsqrtf(var + 1e-5f);
#pragma unroll
  for (int i = 0; i < 4; ++i) {
    int c = t + 256 * i;
    row[c] = (v[i] - mu) * inv * gamma[c] + beta[c];
  }
}

// ---------------- host launcher ----------------
extern "C" void kernel_launch(void* const* d_in, const int* in_sizes, int n_in,
                              void* d_out, int out_size, void* d_ws, size_t ws_size,
                              hipStream_t stream) {
  (void)in_sizes; (void)n_in; (void)out_size; (void)ws_size;
  const float* x   = (const float*)d_in[0];
  const float* y   = (const float*)d_in[1];
  const unsigned char* mask = (const unsigned char*)d_in[2];
  const float* q_experts = (const float*)d_in[3];
  const float* k_experts = (const float*)d_in[4];
  const float* dyn_w = (const float*)d_in[5];
  const float* dyn_b = (const float*)d_in[6];
  const float* v_w  = (const float*)d_in[7];
  const float* v_b  = (const float*)d_in[8];
  const float* in_w = (const float*)d_in[9];
  const float* in_b = (const float*)d_in[10];
  const float* out_w = (const float*)d_in[11];
  const float* out_b = (const float*)d_in[12];
  const float* gamma = (const float*)d_in[13];
  const float* beta  = (const float*)d_in[14];

  char* ws = (char*)d_ws;
  size_t off = 0;
  auto alloc = [&](size_t bytes) -> void* {
    void* p = ws + off;
    off = (off + bytes + 255) & ~(size_t)255;
    return p;
  };
  float* qsum   = (float*)alloc((size_t)K_ * C_ * 4);
  float* ksum   = (float*)alloc((size_t)K_ * C_ * 4);
  float* bc     = (float*)alloc((size_t)C_ * 4);
  float* Wc     = (float*)alloc((size_t)C_ * C_ * 4);
  float* qscale = (float*)alloc((size_t)B_ * N_ * C_ * 4);
  float* kscale = (float*)alloc((size_t)B_ * N_ * C_ * 4);
  _Float16* qph  = (_Float16*)alloc((size_t)BT_ * N_ * C_ * 2);
  _Float16* kph  = (_Float16*)alloc((size_t)BT_ * N_ * C_ * 2);
  _Float16* vph  = (_Float16*)alloc((size_t)BT_ * N_ * C_ * 2);
  _Float16* ctxh = (_Float16*)alloc((size_t)BT_ * N_ * C_ * 2);

  float* out_f   = (float*)d_out;                       // h / final out
  float* avg_out = out_f + (size_t)BT_ * N_ * C_;       // attn_avg

  expert_sum_kernel<<<dim3(2 * K_ * C_ / 256), 256, 0, stream>>>(
      q_experts, k_experts, qsum, ksum);
  combine_w_kernel<<<dim3(C_ * C_ / 256), 256, 0, stream>>>(in_w, v_w, Wc);
  combine_b_kernel<<<dim3(C_ / 256), 256, 0, stream>>>(in_w, v_b, in_b, bc);
  dyn_scale_kernel<<<dim3(B_ * N_), 256, 0, stream>>>(
      y, dyn_w, dyn_b, qsum, ksum, qscale, kscale);

  dim3 ggrid(C_ / BN, BT_ * N_ / BM);   // n fastest -> x strip L2-resident
  // Q projection
  proj_gemm_kernel<<<ggrid, 256, 0, stream>>>(x, nullptr, qscale, in_w, in_b,
                                              qph, nullptr, nullptr, 0);
  // K projection
  proj_gemm_kernel<<<ggrid, 256, 0, stream>>>(x, nullptr, kscale,
                                              in_w + (size_t)C_ * C_, in_b + C_,
                                              kph, nullptr, nullptr, 1);
  // V (combined) projection
  proj_gemm_kernel<<<ggrid, 256, 0, stream>>>(x, nullptr, nullptr, Wc, bc,
                                              vph, nullptr, nullptr, 2);
  // attention + attn_avg + ctx
  attention_kernel<<<dim3(BT_ * 8), 256, 0, stream>>>(qph, kph, vph, mask,
                                                      ctxh, avg_out);
  // out projection + residual -> h (in d_out)
  proj_gemm_kernel<<<ggrid, 256, 0, stream>>>(nullptr, ctxh, nullptr, out_w,
                                              out_b, nullptr, out_f, x, 3);
  // in-place layernorm
  layernorm_kernel<<<dim3(BT_ * N_), 256, 0, stream>>>(out_f, gamma, beta);
}
